// HybridReservoirBlock_58213986730251
// MI455X (gfx1250) — matrix operand order repaired and checked by reading, not compiled, not run on hardware
//
#include <hip/hip_runtime.h>
#include <math.h>

typedef __attribute__((ext_vector_type(2)))  float    v2f;
typedef __attribute__((ext_vector_type(8)))  float    v8f;
typedef __attribute__((ext_vector_type(8)))  _Float16 v8h;
typedef __attribute__((ext_vector_type(16))) _Float16 v16h;

#define BSZ 8
#define SEQ 2048
#define DIM 512
#define HID 1024
#define MROWS (BSZ * SEQ)

// ---------------------------------------------------------------- helpers
static __device__ __forceinline__ float block_reduce_sum(float v, float* sbuf) {
  #pragma unroll
  for (int o = 16; o > 0; o >>= 1) v += __shfl_xor(v, o, 32);
  const int lane = threadIdx.x & 31;
  const int w = threadIdx.x >> 5;
  const int nw = (blockDim.x + 31) >> 5;
  if (lane == 0) sbuf[w] = v;
  __syncthreads();
  v = (threadIdx.x < nw) ? sbuf[threadIdx.x] : 0.0f;
  if (w == 0) {
    #pragma unroll
    for (int o = 16; o > 0; o >>= 1) v += __shfl_xor(v, o, 32);
    if (lane == 0) sbuf[0] = v;
  }
  __syncthreads();
  v = sbuf[0];
  __syncthreads();
  return v;
}

static __device__ __forceinline__ float gelu_exact(float v) {
  return 0.5f * v * (1.0f + erff(v * 0.70710678118f));
}
static __device__ __forceinline__ float sigmoidf(float v) {
  return 1.0f / (1.0f + expf(-v));
}

static __device__ __forceinline__ v16h cat8(v8h lo, v8h hi) {
  return __builtin_shufflevector(lo, hi, 0, 1, 2, 3, 4, 5, 6, 7,
                                 8, 9, 10, 11, 12, 13, 14, 15);
}

#define WMMA_F16(a, b, c)                                                  \
  __builtin_amdgcn_wmma_f32_16x16x32_f16(false, (a), false, (b), (short)0, \
                                         (c), false, false)

// ------------------------------------------------- f32 -> f16 conversion
__global__ void k_f2h(const float* __restrict__ in, _Float16* __restrict__ out,
                      int n) {
  for (int i = blockIdx.x * blockDim.x + threadIdx.x; i < n;
       i += gridDim.x * blockDim.x)
    out[i] = (_Float16)in[i];
}

// ------------------------------------------------- LN over D + delta scalar
__global__ void k_ln_x(const float* __restrict__ x, const float* __restrict__ g,
                       const float* __restrict__ b, const float* __restrict__ wd,
                       const float* __restrict__ bd, _Float16* __restrict__ xnh,
                       float* __restrict__ delta) {
  __shared__ float sbuf[32];
  const int row = blockIdx.x;
  const float* xr = x + (size_t)row * DIM;
  _Float16* onr = xnh + (size_t)row * DIM;
  const int i0 = threadIdx.x * 2;
  const float x0 = xr[i0], x1 = xr[i0 + 1];
  const float s  = block_reduce_sum(x0 + x1, sbuf);
  const float sq = block_reduce_sum(x0 * x0 + x1 * x1, sbuf);
  const float mean = s * (1.0f / DIM);
  const float var  = sq * (1.0f / DIM) - mean * mean;
  const float rstd = rsqrtf(var + 1e-5f);
  const float n0 = (x0 - mean) * rstd * g[i0] + b[i0];
  const float n1 = (x1 - mean) * rstd * g[i0 + 1] + b[i0 + 1];
  onr[i0] = (_Float16)n0;
  onr[i0 + 1] = (_Float16)n1;
  const float dd = block_reduce_sum(n0 * wd[i0] + n1 * wd[i0 + 1], sbuf);
  if (threadIdx.x == 0) {
    const float z = dd + bd[0];
    delta[row] = (z > 20.0f) ? z : log1pf(expf(z));  // softplus
  }
}

// ------------------------------------------------- LN over H=1024 -> f16
__global__ void k_ln_h(const float* __restrict__ in, _Float16* __restrict__ outh,
                       const float* __restrict__ g, const float* __restrict__ b) {
  __shared__ float sbuf[32];
  const int row = blockIdx.x;
  const float* r = in + (size_t)row * HID;
  _Float16* o = outh + (size_t)row * HID;
  float v[4];
  float s = 0.0f, sq = 0.0f;
  #pragma unroll
  for (int j = 0; j < 4; ++j) {
    v[j] = r[threadIdx.x + j * 256];
    s += v[j];
    sq += v[j] * v[j];
  }
  s  = block_reduce_sum(s, sbuf);
  sq = block_reduce_sum(sq, sbuf);
  const float mean = s * (1.0f / HID);
  const float rstd = rsqrtf(sq * (1.0f / HID) - mean * mean + 1e-5f);
  #pragma unroll
  for (int j = 0; j < 4; ++j) {
    const int i = threadIdx.x + j * 256;
    o[i] = (_Float16)((v[j] - mean) * rstd * g[i] + b[i]);
  }
}

// ------------------------------------------------- f16 WMMA GEMM (16x16x32)
// C[M,N] = A[M,K] @ W[N,K]^T ; f32 accumulate; 16x32 tile per wave.
__global__ void k_gemm_h(const _Float16* __restrict__ A,
                         const _Float16* __restrict__ W,
                         float* __restrict__ C, int Mr, int K, int N) {
  const int lane = threadIdx.x & 31;
  const int wave = threadIdx.x >> 5;
  const int lm = lane & 15;
  const int hi = lane >> 4;
  const int tn_count = N >> 5;
  const int tile = blockIdx.x * (blockDim.x >> 5) + wave;
  const int tm = tile / tn_count;
  const int tn = tile % tn_count;
  if (tm * 16 >= Mr) return;
  const _Float16* Ar = A + (size_t)(tm * 16 + lm) * K;
  const _Float16* W0 = W + (size_t)(tn * 32 + lm) * K;
  const _Float16* W1 = W + (size_t)(tn * 32 + 16 + lm) * K;
  const int aoff = hi * 16;
  const int boff = hi * 16;
  v8f c0 = {0.f, 0.f, 0.f, 0.f, 0.f, 0.f, 0.f, 0.f};
  v8f c1 = {0.f, 0.f, 0.f, 0.f, 0.f, 0.f, 0.f, 0.f};
  #pragma unroll 2
  for (int k = 0; k < K; k += 32) {
    __builtin_prefetch(Ar + k + 128, 0, 3);
    __builtin_prefetch(W0 + k + 128, 0, 3);
    __builtin_prefetch(W1 + k + 128, 0, 3);
    const v16h a = cat8(*(const v8h*)(Ar + k + aoff),
                        *(const v8h*)(Ar + k + aoff + 8));
    const v16h b0 = cat8(*(const v8h*)(W0 + k + boff),
                         *(const v8h*)(W0 + k + boff + 8));
    const v16h b1 = cat8(*(const v8h*)(W1 + k + boff),
                         *(const v8h*)(W1 + k + boff + 8));
    c0 = WMMA_F16(a, b0, c0);
    c1 = WMMA_F16(a, b1, c1);
  }
  const int col0 = tn * 32 + lm;
  const int col1 = col0 + 16;
  #pragma unroll
  for (int r = 0; r < 8; ++r) {
    const int m = tm * 16 + r + hi * 8;
    C[(size_t)m * N + col0] = c0[r];
    C[(size_t)m * N + col1] = c1[r];
  }
}

// ------------------------------------------------- fused readout + gate + mix
__global__ void k_readout_fused(const _Float16* __restrict__ rsh,
                                const _Float16* __restrict__ ssh,
                                const _Float16* __restrict__ xh,
                                const _Float16* __restrict__ wro,
                                const _Float16* __restrict__ wg,
                                const float* __restrict__ b_ro,
                                const float* __restrict__ b_gate,
                                float* __restrict__ out) {
  const int lane = threadIdx.x & 31;
  const int wave = threadIdx.x >> 5;
  const int lm = lane & 15;
  const int hi = lane >> 4;
  const int tn_count = DIM >> 5;  // 16
  const int tile = blockIdx.x * (blockDim.x >> 5) + wave;
  const int tm = tile / tn_count;
  const int tn = tile % tn_count;
  const int aoff = hi * 16;
  const int boff = hi * 16;

  // --- readout GEMMs over K = HID, sharing W_ro fragments ---
  const _Float16* Ar = rsh + (size_t)(tm * 16 + lm) * HID;
  const _Float16* As = ssh + (size_t)(tm * 16 + lm) * HID;
  const _Float16* W0 = wro + (size_t)(tn * 32 + lm) * HID;
  const _Float16* W1 = wro + (size_t)(tn * 32 + 16 + lm) * HID;
  v8f cr0 = {0.f, 0.f, 0.f, 0.f, 0.f, 0.f, 0.f, 0.f};
  v8f cr1 = cr0, cs0 = cr0, cs1 = cr0;
  #pragma unroll 2
  for (int k = 0; k < HID; k += 32) {
    __builtin_prefetch(Ar + k + 128, 0, 3);
    __builtin_prefetch(As + k + 128, 0, 3);
    const v16h ar = cat8(*(const v8h*)(Ar + k + aoff),
                         *(const v8h*)(Ar + k + aoff + 8));
    const v16h as = cat8(*(const v8h*)(As + k + aoff),
                         *(const v8h*)(As + k + aoff + 8));
    const v16h b0 = cat8(*(const v8h*)(W0 + k + boff),
                         *(const v8h*)(W0 + k + boff + 8));
    const v16h b1 = cat8(*(const v8h*)(W1 + k + boff),
                         *(const v8h*)(W1 + k + boff + 8));
    cr0 = WMMA_F16(ar, b0, cr0);
    cs0 = WMMA_F16(as, b0, cs0);
    cr1 = WMMA_F16(ar, b1, cr1);
    cs1 = WMMA_F16(as, b1, cs1);
  }

  // --- gate GEMM over K = DIM ---
  const _Float16* Ag = xh + (size_t)(tm * 16 + lm) * DIM;
  const _Float16* G0 = wg + (size_t)(tn * 32 + lm) * DIM;
  const _Float16* G1 = wg + (size_t)(tn * 32 + 16 + lm) * DIM;
  v8f cg0 = {0.f, 0.f, 0.f, 0.f, 0.f, 0.f, 0.f, 0.f};
  v8f cg1 = cg0;
  #pragma unroll 2
  for (int k = 0; k < DIM; k += 32) {
    const v16h ag = cat8(*(const v8h*)(Ag + k + aoff),
                         *(const v8h*)(Ag + k + aoff + 8));
    const v16h g0 = cat8(*(const v8h*)(G0 + k + boff),
                         *(const v8h*)(G0 + k + boff + 8));
    const v16h g1 = cat8(*(const v8h*)(G1 + k + boff),
                         *(const v8h*)(G1 + k + boff + 8));
    cg0 = WMMA_F16(ag, g0, cg0);
    cg1 = WMMA_F16(ag, g1, cg1);
  }

  // --- epilogue: act + gated lerp, single store of d_out ---
  const int col0 = tn * 32 + lm;
  const int col1 = col0 + 16;
  const float bro0 = b_ro[col0], bro1 = b_ro[col1];
  const float bg0 = b_gate[col0], bg1 = b_gate[col1];
  #pragma unroll
  for (int r = 0; r < 8; ++r) {
    const int m = tm * 16 + r + hi * 8;
    const float g0v = sigmoidf(cg0[r] + bg0);
    const float g1v = sigmoidf(cg1[r] + bg1);
    const float r0 = gelu_exact(cr0[r] + bro0);
    const float r1 = gelu_exact(cr1[r] + bro1);
    const float s0 = gelu_exact(cs0[r] + bro0);
    const float s1 = gelu_exact(cs1[r] + bro1);
    out[(size_t)m * DIM + col0] = r0 * g0v + s0 * (1.0f - g0v);
    out[(size_t)m * DIM + col1] = r1 * g1v + s1 * (1.0f - g1v);
  }
}

// ------------------------------------------------- reservoir recurrence scan
// fp32 WMMA 16x16x4; 8 blocks x 256 threads (64 waves) persistent over SEQ
// steps; h_prev staged into LDS via GLOBAL_LOAD_ASYNC_TO_LDS_B128 (ASYNCcnt),
// W_res streamed from L2, grid barrier per step.
#define LDSSTRIDE 1028  // 4112 B row stride: 16B-aligned rows, banks (4L)%64
__global__ void k_res_scan(const float* __restrict__ xin,
                           const float* __restrict__ Wres,
                           float* __restrict__ hbuf, float* __restrict__ states,
                           int* __restrict__ sync) {
  __shared__ float lds[BSZ * LDSSTRIDE];
  const int lane = threadIdx.x & 31;
  const int wave = threadIdx.x >> 5;
  const int lm = lane & 15;
  const int hi = lane >> 4;
  const int koff = hi * 2;
  const int tile = blockIdx.x * 8 + wave;
  const int n0 = tile * 16;
  const int col = n0 + lm;
  const int nblocks = (int)gridDim.x;
  const float* Wr = Wres + (size_t)(n0 + lm) * HID + koff;
  const bool avalid = (lm < BSZ);
  const unsigned ldsbase = (unsigned)(size_t)(&lds[0]);

  for (int t = 0; t < SEQ; ++t) {
    const float* cur = hbuf + (t & 1) * (BSZ * HID);
    float* nxt = hbuf + ((t + 1) & 1) * (BSZ * HID);

    // async copy h_prev (8 x 4096 B) straight into LDS, 16 B per lane per op
    #pragma unroll
    for (int j = 0; j < (BSZ * HID * 4) / (16 * 256); ++j) {  // 8 issues/thread
      const int id = (int)threadIdx.x + j * 256;  // 16B chunk id
      const int row = id >> 8;                    // 256 chunks per 4096B row
      const int inrow = (id & 255) * 16;          // byte offset within row
      const unsigned laddr = ldsbase + (unsigned)(row * (LDSSTRIDE * 4) + inrow);
      const float* gaddr = cur + (size_t)row * HID + (inrow >> 2);
      asm volatile("global_load_async_to_lds_b128 %0, %1, off"
                   :: "v"(laddr), "v"(gaddr)
                   : "memory");
    }

    v8f c;
    #pragma unroll
    for (int r = 0; r < 8; ++r) {
      const int bb = r + hi * 8;
      c[r] = (bb < BSZ) ? xin[((size_t)bb * SEQ + t) * HID + col] : 0.0f;
    }
    // warm next step's xin rows while we compute (per-lane prefetch)
    if (t + 1 < SEQ)
      __builtin_prefetch(&xin[((size_t)(lane & 7) * SEQ + (t + 1)) * HID + n0],
                         0, 3);

    asm volatile("s_wait_asynccnt 0x0" ::: "memory");
    __syncthreads();

    const float* Ar = &lds[lm * LDSSTRIDE + koff];
    #pragma unroll 4
    for (int k = 0; k < HID; k += 4) {
      v2f a;
      if (avalid) a = *(const v2f*)(Ar + k);
      else { a.x = 0.0f; a.y = 0.0f; }
      const v2f bw = *(const v2f*)(Wr + k);
      c = __builtin_amdgcn_wmma_f32_16x16x4_f32(false, a, false, bw, (short)0, c,
                                                false, false);
    }

    if (hi == 0) {
      #pragma unroll
      for (int r = 0; r < 8; ++r) {
        const float prev = lds[r * LDSSTRIDE + col];
        const float nh = 0.95f * tanhf(c[r]) + 0.05f * prev;
        nxt[r * HID + col] = nh;
        states[((size_t)r * SEQ + t) * HID + col] = nh;
      }
    }
    __threadfence();
    __syncthreads();
    if (threadIdx.x == 0) {
      const int gen =
          __hip_atomic_load(sync + 1, __ATOMIC_ACQUIRE, __HIP_MEMORY_SCOPE_AGENT);
      if (__hip_atomic_fetch_add(sync, 1, __ATOMIC_ACQ_REL,
                                 __HIP_MEMORY_SCOPE_AGENT) == nblocks - 1) {
        __hip_atomic_store(sync, 0, __ATOMIC_RELAXED, __HIP_MEMORY_SCOPE_AGENT);
        __hip_atomic_store(sync + 1, gen + 1, __ATOMIC_RELEASE,
                           __HIP_MEMORY_SCOPE_AGENT);
      } else {
        while (__hip_atomic_load(sync + 1, __ATOMIC_ACQUIRE,
                                 __HIP_MEMORY_SCOPE_AGENT) == gen)
          __builtin_amdgcn_s_sleep(1);
      }
    }
    __syncthreads();
  }
}

// ------------------------------------------------- diagonal SSM scan, in place
__global__ void k_ssm_scan(float* __restrict__ bm, const float* __restrict__ delta,
                           const float* __restrict__ Aarr) {
  const int gid = blockIdx.x * blockDim.x + threadIdx.x;  // 0..8191
  const int b = gid >> 10;
  const int h = gid & 1023;
  const float Av = Aarr[h];
  const float inv = 1.0f / (Av + 1e-8f);
  const float* dl = delta + (size_t)b * SEQ;
  float* p = bm + (size_t)b * SEQ * HID + h;
  float hst = 0.0f;
  for (int s = 0; s < SEQ; ++s) {
    const float ab = expf(Av * dl[s]);
    const float bb = p[(size_t)s * HID] * (1.0f - ab) * inv;
    hst = ab * hst + bb;
    p[(size_t)s * HID] = hst;
  }
}

// ------------------------------------------------- init
__global__ void k_init(float* __restrict__ hbuf, int* __restrict__ sync) {
  const int i = blockIdx.x * blockDim.x + threadIdx.x;
  if (i < 2 * BSZ * HID) hbuf[i] = 0.0f;
  if (i < 2) sync[i] = 0;
}

// ---------------------------------------------------------------- launcher
extern "C" void kernel_launch(void* const* d_in, const int* in_sizes, int n_in,
                              void* d_out, int out_size, void* d_ws, size_t ws_size,
                              hipStream_t stream) {
  (void)in_sizes; (void)n_in; (void)out_size; (void)ws_size;
  const float* x      = (const float*)d_in[0];
  const float* norm_g = (const float*)d_in[1];
  const float* norm_b = (const float*)d_in[2];
  const float* W_in   = (const float*)d_in[3];
  const float* W_res  = (const float*)d_in[4];
  const float* A      = (const float*)d_in[5];
  const float* W_B    = (const float*)d_in[6];
  const float* w_del  = (const float*)d_in[7];
  const float* b_del  = (const float*)d_in[8];
  const float* ro_g   = (const float*)d_in[9];
  const float* ro_b   = (const float*)d_in[10];
  const float* W_ro   = (const float*)d_in[11];
  const float* b_ro   = (const float*)d_in[12];
  const float* W_gate = (const float*)d_in[13];
  const float* b_gate = (const float*)d_in[14];
  float* out = (float*)d_out;

  // workspace layout
  char* wsb = (char*)d_ws;
  size_t off = 0;
  auto alloc = [&](size_t bytes) {
    void* p = wsb + off;
    off += (bytes + 255) & ~(size_t)255;
    return p;
  };
  float*    xin   = (float*)alloc((size_t)MROWS * HID * 4);   // res input proj
  float*    bm    = (float*)alloc((size_t)MROWS * HID * 4);   // Bm -> ssm states
  float*    rs    = (float*)alloc((size_t)MROWS * HID * 4);   // res states
  _Float16* xnh   = (_Float16*)alloc((size_t)MROWS * DIM * 2);  // LN(x) f16
  _Float16* xh    = (_Float16*)alloc((size_t)MROWS * DIM * 2);  // raw x f16
  _Float16* rsh   = (_Float16*)alloc((size_t)MROWS * HID * 2);  // LN(res) f16
  _Float16* ssh   = (_Float16*)alloc((size_t)MROWS * HID * 2);  // LN(ssm) f16
  _Float16* winh  = (_Float16*)alloc((size_t)HID * DIM * 2);
  _Float16* wbh   = (_Float16*)alloc((size_t)HID * DIM * 2);
  _Float16* wgh   = (_Float16*)alloc((size_t)DIM * DIM * 2);
  _Float16* wroh  = (_Float16*)alloc((size_t)DIM * HID * 2);
  float*    delta = (float*)alloc((size_t)MROWS * 4);
  float*    hbuf  = (float*)alloc((size_t)2 * BSZ * HID * 4);
  int*      syncp = (int*)alloc(256);

  k_init<<<64, 256, 0, stream>>>(hbuf, syncp);

  // f32 -> f16 conversions (weights + raw x for the gate)
  k_f2h<<<2048, 256, 0, stream>>>(x, xh, MROWS * DIM);
  k_f2h<<<512, 256, 0, stream>>>(W_in, winh, HID * DIM);
  k_f2h<<<512, 256, 0, stream>>>(W_B, wbh, HID * DIM);
  k_f2h<<<256, 256, 0, stream>>>(W_gate, wgh, DIM * DIM);
  k_f2h<<<512, 256, 0, stream>>>(W_ro, wroh, DIM * HID);

  // LN(x) -> f16 + delta
  k_ln_x<<<MROWS, 256, 0, stream>>>(x, norm_g, norm_b, w_del, b_del, xnh, delta);

  // xin = LN(x) @ W_in^T ; Bm = LN(x) @ W_B^T   (K=512, N=1024)
  k_gemm_h<<<(MROWS / 16) * (HID / 32) / 8, 256, 0, stream>>>(
      xnh, winh, xin, MROWS, DIM, HID);
  k_gemm_h<<<(MROWS / 16) * (HID / 32) / 8, 256, 0, stream>>>(
      xnh, wbh, bm, MROWS, DIM, HID);

  // reservoir scan (persistent, grid-synced, fp32 WMMA, async-LDS staging)
  k_res_scan<<<8, 256, 0, stream>>>(xin, W_res, hbuf, rs, syncp);

  // diagonal SSM scan (in-place over bm)
  k_ssm_scan<<<(BSZ * HID) / 256, 256, 0, stream>>>(bm, delta, A);

  // readout LN -> f16
  k_ln_h<<<MROWS, 256, 0, stream>>>(rs, rsh, ro_g, ro_b);
  k_ln_h<<<MROWS, 256, 0, stream>>>(bm, ssh, ro_g, ro_b);

  // fused: both readout GEMMs (shared W_ro frags) + gate GEMM + mix -> d_out
  k_readout_fused<<<(MROWS / 16) * (DIM / 32) / 8, 256, 0, stream>>>(
      rsh, ssh, xh, wroh, wgh, b_ro, b_gate, out);
}
